// AttentionModule_53463752900893
// MI455X (gfx1250) — matrix operand order repaired and checked by reading, not compiled, not run on hardware
//
#include <hip/hip_runtime.h>

typedef __attribute__((ext_vector_type(16))) _Float16 v16h;
typedef __attribute__((ext_vector_type(8)))  float    v8f;

// D = A(16x32 f16) * B(32x16 f16) + C(16x16 f32), wave32
#define WMMA_F16(a, b, c) \
  __builtin_amdgcn_wmma_f32_16x16x32_f16(false, (a), false, (b), (short)0, (c), false, false)

constexpr int BB   = 8;
constexpr int HH   = 64;
constexpr int WW   = 64;
constexpr int CC   = 256;
constexpr int KEY  = 32;
constexpr int NPIX = HH * WW;     // 4096
constexpr int BN   = BB * NPIX;   // 32768

// ---------------------------------------------------------------------------
// A fragment (16x32, f16) from a row-major f16 matrix, k-chunk at kbase.
// Lane l: m = l&15, hi = l>>4. Halves 0..7: k = hi*8 + j ; 8..15: k = 16+hi*8+j
// ---------------------------------------------------------------------------
__device__ inline v16h a_frag(const _Float16* row, int kbase, int hi) {
  v16h a;
#pragma unroll
  for (int j = 0; j < 8; ++j) {
    a[j]     = row[kbase + hi * 8 + j];
    a[8 + j] = row[kbase + 16 + hi * 8 + j];
  }
  return a;
}

// ---------------------------------------------------------------------------
// B fragment (32x16, f16) from a k-contiguous (transposed) matrix.
// Lane l: n = l&15, hi = l>>4. Halves 0..7: k = hi*8 + j ; 8..15: k = 16+hi*8+j
// col = pointer to column n's k-run start.
// ---------------------------------------------------------------------------
__device__ inline v16h b_frag(const _Float16* col, int hi) {
  v16h b;
#pragma unroll
  for (int j = 0; j < 8; ++j) {
    b[j]     = col[hi * 8 + j];
    b[8 + j] = col[16 + hi * 8 + j];
  }
  return b;
}

// ===========================================================================
// Kernel 1: weights f32 -> f16, transposed (WT[n][k] = W[k][n])
// ===========================================================================
__global__ void prep_weights(const float* __restrict__ Wf, const float* __restrict__ Wg,
                             const float* __restrict__ Wh, const float* __restrict__ Wo,
                             _Float16* __restrict__ WfT, _Float16* __restrict__ WgT,
                             _Float16* __restrict__ WhT, _Float16* __restrict__ WoT) {
  int i = blockIdx.x * blockDim.x + threadIdx.x;
  if (i < CC * KEY) {
    int k = i / KEY, n = i % KEY;
    WfT[n * CC + k] = (_Float16)Wf[i];
    WgT[n * CC + k] = (_Float16)Wg[i];
  }
  if (i < CC * CC) {
    int k = i / CC, n = i % CC;
    WhT[n * CC + k] = (_Float16)Wh[i];
    WoT[n * CC + k] = (_Float16)Wo[i];
  }
}

// ===========================================================================
// Kernel 2: QKV projection. One wave = 16 pixels. x row tile cached as 8
// A-fragments; Q,K row-major f16 [row][32]; V transposed f16 [b][c][pix].
// ===========================================================================
__global__ __launch_bounds__(128) void proj_qkv(
    const float* __restrict__ x,
    const _Float16* __restrict__ WfT, const float* __restrict__ bf,
    const _Float16* __restrict__ WgT, const float* __restrict__ bg,
    const _Float16* __restrict__ WhT, const float* __restrict__ bh,
    _Float16* __restrict__ Q, _Float16* __restrict__ K, _Float16* __restrict__ Vt) {
  const int wave = (blockIdx.x * blockDim.x + threadIdx.x) >> 5;
  const int lane = threadIdx.x & 31;
  const int m = lane & 15, hi = lane >> 4, n = lane & 15;
  const long p0 = (long)wave * 16;

  // Cache the 16x256 x tile as f16 A-fragments (convert f32->f16 on load).
  v16h a[8];
  const float* xrow = x + (p0 + m) * CC;
#pragma unroll
  for (int kc = 0; kc < 8; ++kc) {
    v16h t;
#pragma unroll
    for (int j = 0; j < 8; ++j) {
      t[j]     = (_Float16)xrow[kc * 32 + hi * 8 + j];
      t[8 + j] = (_Float16)xrow[kc * 32 + 16 + hi * 8 + j];
    }
    a[kc] = t;
  }

  // Q and K: 2 column tiles each (32 outputs).
#pragma unroll
  for (int ct = 0; ct < 2; ++ct) {
    v8f accq = {}; v8f acck = {};
#pragma unroll
    for (int kc = 0; kc < 8; ++kc) {
      v16h bq = b_frag(WfT + (ct * 16 + n) * CC + kc * 32, hi);
      v16h bk = b_frag(WgT + (ct * 16 + n) * CC + kc * 32, hi);
      accq = WMMA_F16(a[kc], bq, accq);
      acck = WMMA_F16(a[kc], bk, acck);
    }
    const float bqv = bf[ct * 16 + n];
    const float bkv = bg[ct * 16 + n];
#pragma unroll
    for (int r = 0; r < 8; ++r) {
      const long row = p0 + r + 8 * hi;
      Q[row * KEY + ct * 16 + n] = (_Float16)(accq[r] + bqv);
      K[row * KEY + ct * 16 + n] = (_Float16)(acck[r] + bkv);
    }
  }

  // V: 16 column tiles, stored transposed [b][c][pix].
  const long bidx = p0 / NPIX;
  const long pix0 = p0 % NPIX;
#pragma unroll 4
  for (int ct = 0; ct < 16; ++ct) {
    v8f acc = {};
#pragma unroll
    for (int kc = 0; kc < 8; ++kc) {
      v16h bv = b_frag(WhT + (ct * 16 + n) * CC + kc * 32, hi);
      acc = WMMA_F16(a[kc], bv, acc);
    }
    const float bias = bh[ct * 16 + n];
#pragma unroll
    for (int r = 0; r < 8; ++r)
      Vt[(bidx * CC + ct * 16 + n) * (long)NPIX + pix0 + r + 8 * hi] =
          (_Float16)(acc[r] + bias);
  }
}

// ===========================================================================
// Kernel 3: flash attention. One wave = one 16-query tile, key blocks of 64.
// Online softmax; scores never leave registers/LDS. O accum: 16x256 f32.
// ===========================================================================
__global__ __launch_bounds__(128) void attention(
    const _Float16* __restrict__ Q, const _Float16* __restrict__ K,
    const _Float16* __restrict__ Vt, _Float16* __restrict__ O) {
  __shared__ _Float16 lds[4][16 * 64];  // per-wave P staging (D->A relayout)
  const int wib  = threadIdx.x >> 5;
  const int wave = blockIdx.x * 4 + wib;
  const int lane = threadIdx.x & 31;
  const int m = lane & 15, hi = lane >> 4, n = lane & 15;
  const int bidx = wave >> 8;    // 256 query tiles per batch
  const int qt   = wave & 255;
  const long qbase = (long)bidx * NPIX + qt * 16;

  // Q A-fragment: K-dim 32 == full head dim, single fragment, reused all loop.
  const v16h aq = a_frag(Q + (qbase + m) * KEY, 0, hi);

  v8f o[16];
  const v8f vzero = {};
#pragma unroll
  for (int t = 0; t < 16; ++t) o[t] = vzero;
  float mrow[8], lrow[8];
#pragma unroll
  for (int r = 0; r < 8; ++r) { mrow[r] = -1e30f; lrow[r] = 0.0f; }

  const _Float16* Kb = K + (long)bidx * NPIX * KEY;
  const _Float16* Vb = Vt + (long)bidx * CC * NPIX;
  _Float16* P = lds[wib];

  for (int kb = 0; kb < NPIX; kb += 64) {
    if (kb + 64 < NPIX)
      __builtin_prefetch(Kb + (long)(kb + 64 + n) * KEY, 0, 1);  // global_prefetch_b8

    // S = Q * K^T for 64 keys: 4 WMMAs (each full K-dim=32).
    v8f s[4];
#pragma unroll
    for (int kt = 0; kt < 4; ++kt) {
      v16h bk = b_frag(Kb + (long)(kb + kt * 16 + n) * KEY, hi);
      s[kt] = WMMA_F16(aq, bk, vzero);
    }

    // Online softmax: per-row (r + 8*hi) max/sum via 16-lane reductions.
    float scale[8];
#pragma unroll
    for (int r = 0; r < 8; ++r) {
      float v = fmaxf(fmaxf(s[0][r], s[1][r]), fmaxf(s[2][r], s[3][r]));
#pragma unroll
      for (int sh = 1; sh < 16; sh <<= 1) v = fmaxf(v, __shfl_xor(v, sh, 32));
      const float mn = fmaxf(mrow[r], v);
      scale[r] = __expf(mrow[r] - mn);
      mrow[r]  = mn;
    }
#pragma unroll
    for (int r = 0; r < 8; ++r) {
      const int row = r + 8 * hi;
      float rs = 0.0f;
#pragma unroll
      for (int kt = 0; kt < 4; ++kt) {
        const float p = __expf(s[kt][r] - mrow[r]);
        rs += p;
        P[row * 64 + kt * 16 + n] = (_Float16)p;  // D layout -> LDS
      }
#pragma unroll
      for (int sh = 1; sh < 16; sh <<= 1) rs += __shfl_xor(rs, sh, 32);
      lrow[r] = lrow[r] * scale[r] + rs;
    }

    // Rescale O accumulators (scale is uniform across the 16-lane half).
#pragma unroll
    for (int t = 0; t < 16; ++t)
#pragma unroll
      for (int r = 0; r < 8; ++r) o[t][r] *= scale[r];

    // Reload P in A layout (two 16x32 fragments). LDS in-order within wave.
    v16h ap[2];
#pragma unroll
    for (int h2 = 0; h2 < 2; ++h2) {
      v16h a;
#pragma unroll
      for (int j = 0; j < 8; ++j) {
        a[j]     = P[m * 64 + h2 * 32 + hi * 8 + j];
        a[8 + j] = P[m * 64 + h2 * 32 + 16 + hi * 8 + j];
      }
      ap[h2] = a;
    }

    // O += P * V : 16 channel tiles x 2 k-chunks = 32 WMMAs per key block.
#pragma unroll 4
    for (int t = 0; t < 16; ++t) {
      v8f acc = o[t];
#pragma unroll
      for (int h2 = 0; h2 < 2; ++h2) {
        v16h bv = b_frag(Vb + (long)(t * 16 + n) * NPIX + kb + h2 * 32, hi);
        acc = WMMA_F16(ap[h2], bv, acc);
      }
      o[t] = acc;
    }
  }

  // Normalize by softmax denominator and store O as f16 row-major [row][256].
#pragma unroll
  for (int r = 0; r < 8; ++r) lrow[r] = 1.0f / lrow[r];
#pragma unroll
  for (int t = 0; t < 16; ++t)
#pragma unroll
    for (int r = 0; r < 8; ++r) {
      const long row = qbase + r + 8 * hi;
      O[row * CC + t * 16 + n] = (_Float16)(o[t][r] * lrow[r]);
    }
}

// ===========================================================================
// Kernel 4: output projection  out = O @ Wo + bo  (f32 out)
// ===========================================================================
__global__ __launch_bounds__(128) void proj_out(
    const _Float16* __restrict__ O, const _Float16* __restrict__ WoT,
    const float* __restrict__ bo, float* __restrict__ out) {
  const int wave = (blockIdx.x * blockDim.x + threadIdx.x) >> 5;
  const int lane = threadIdx.x & 31;
  const int m = lane & 15, hi = lane >> 4, n = lane & 15;
  const long p0 = (long)wave * 16;

  v16h a[8];
  const _Float16* row = O + (p0 + m) * CC;
#pragma unroll
  for (int kc = 0; kc < 8; ++kc) a[kc] = a_frag(row, kc * 32, hi);

#pragma unroll 4
  for (int ct = 0; ct < 16; ++ct) {
    v8f acc = {};
#pragma unroll
    for (int kc = 0; kc < 8; ++kc) {
      v16h bw = b_frag(WoT + (ct * 16 + n) * CC + kc * 32, hi);
      acc = WMMA_F16(a[kc], bw, acc);
    }
    const float bias = bo[ct * 16 + n];
#pragma unroll
    for (int r = 0; r < 8; ++r)
      out[(p0 + r + 8 * hi) * CC + ct * 16 + n] = acc[r] + bias;
  }
}

// ===========================================================================
extern "C" void kernel_launch(void* const* d_in, const int* in_sizes, int n_in,
                              void* d_out, int out_size, void* d_ws, size_t ws_size,
                              hipStream_t stream) {
  (void)in_sizes; (void)n_in; (void)out_size; (void)ws_size;
  const float* x  = (const float*)d_in[0];
  const float* Wf = (const float*)d_in[1];
  const float* bf = (const float*)d_in[2];
  const float* Wg = (const float*)d_in[3];
  const float* bg = (const float*)d_in[4];
  const float* Wh = (const float*)d_in[5];
  const float* bh = (const float*)d_in[6];
  const float* Wo = (const float*)d_in[7];
  const float* bo = (const float*)d_in[8];

  char* ws = (char*)d_ws;
  size_t off = 0;
  auto alloc = [&](size_t bytes) -> void* {
    void* p = ws + off;
    off += (bytes + 255) & ~(size_t)255;
    return p;
  };
  _Float16* WfT = (_Float16*)alloc((size_t)KEY * CC * 2);
  _Float16* WgT = (_Float16*)alloc((size_t)KEY * CC * 2);
  _Float16* WhT = (_Float16*)alloc((size_t)CC * CC * 2);
  _Float16* WoT = (_Float16*)alloc((size_t)CC * CC * 2);
  _Float16* Qh  = (_Float16*)alloc((size_t)BN * KEY * 2);
  _Float16* Kh  = (_Float16*)alloc((size_t)BN * KEY * 2);
  _Float16* Vt  = (_Float16*)alloc((size_t)BN * CC * 2);   // [b][c][pix]
  _Float16* Oh  = (_Float16*)alloc((size_t)BN * CC * 2);   // [row][c]

  prep_weights<<<(CC * CC + 255) / 256, 256, 0, stream>>>(Wf, Wg, Wh, Wo, WfT, WgT, WhT, WoT);
  proj_qkv<<<BN / 16 / 4, 128, 0, stream>>>(x, WfT, bf, WgT, bg, WhT, bh, Qh, Kh, Vt);
  attention<<<(BB * NPIX / 16) / 4, 128, 0, stream>>>(Qh, Kh, Vt, Oh);
  proj_out<<<BN / 16 / 4, 128, 0, stream>>>(Oh, WoT, bo, (float*)d_out);
}
